// SAFIN_25975962206257
// MI455X (gfx1250) — matrix operand order repaired and checked by reading, not compiled, not run on hardware
//
#include <hip/hip_runtime.h>

// ---------------------------------------------------------------------------
// SAFIN on gfx1250 (MI455X): bf16 WMMA everywhere, flash-attention streaming
// softmax (wave-parallel, shuffle-reduced), async-to-LDS double-buffered GEMM,
// LDS-tiled transposes, direct-load WMMA fragments.
// ---------------------------------------------------------------------------

typedef __attribute__((ext_vector_type(16))) __bf16 v16bf;
typedef __attribute__((ext_vector_type(8)))  float  v8f;

struct __align__(16) U4 { unsigned int x, y, z, w; };
union Frag { v16bf v; U4 u[2]; };

#define WMMA_BF16(a, b, c) \
    __builtin_amdgcn_wmma_f32_16x16x32_bf16(false, (a), false, (b), (short)0, (c), false, false)

#if defined(__AMDGCN__) && __has_builtin(__builtin_amdgcn_global_load_async_to_lds_b128)
#define USE_ASYNC_LDS 1
typedef int v4i __attribute__((vector_size(16)));
typedef __attribute__((address_space(1))) v4i as1_v4i;   // global v4i
typedef __attribute__((address_space(3))) v4i as3_v4i;   // LDS v4i
#define ASYNC_B128(gptr, lptr)                                               \
    __builtin_amdgcn_global_load_async_to_lds_b128(                          \
        (as1_v4i*)(unsigned long long)(gptr),                                \
        (as3_v4i*)(unsigned int)(unsigned long long)(lptr), 0, 0)
#else
#define USE_ASYNC_LDS 0
#endif

static __device__ __forceinline__ void wait_async0()
{
#if __has_builtin(__builtin_amdgcn_s_wait_asynccnt)
    __builtin_amdgcn_s_wait_asynccnt(0);
#else
    asm volatile("s_wait_asynccnt 0" ::: "memory");
#endif
}

static constexpr int Cc  = 512;    // channels
static constexpr int NSP = 4096;   // H*W
static constexpr int Bb  = 4;      // batch

// ---------------------------------------------------------------------------
// 1) Per-(src,b,c) mean / rstd over 4096 spatial elements.
// ---------------------------------------------------------------------------
__global__ __launch_bounds__(256) void stats_kernel(
    const float* __restrict__ content, const float* __restrict__ style,
    float* __restrict__ stats)
{
    const int c = blockIdx.x, b = blockIdx.y, src = blockIdx.z;
    const float* base = (src == 0 ? content : style) + ((size_t)b * Cc + c) * NSP;
    float s = 0.f, ss = 0.f;
    for (int i = threadIdx.x; i < NSP; i += 256) {
        float x = base[i];
        s += x; ss += x * x;
    }
    __shared__ float r1[256], r2[256];
    r1[threadIdx.x] = s; r2[threadIdx.x] = ss;
    __syncthreads();
    for (int o = 128; o > 0; o >>= 1) {
        if (threadIdx.x < o) { r1[threadIdx.x] += r1[threadIdx.x + o];
                               r2[threadIdx.x] += r2[threadIdx.x + o]; }
        __syncthreads();
    }
    if (threadIdx.x == 0) {
        float mean = r1[0] * (1.f / NSP);
        float var  = r2[0] * (1.f / NSP) - mean * mean;
        ((float2*)stats)[(src * Bb + b) * Cc + c] = make_float2(mean, rsqrtf(var + 1e-5f));
    }
}

// ---------------------------------------------------------------------------
// 2) fp32 -> bf16 weight conversion.
// ---------------------------------------------------------------------------
__global__ void cvt_bf16_kernel(const float* __restrict__ src, __bf16* __restrict__ dst, int n)
{
    int i = blockIdx.x * 256 + threadIdx.x;
    if (i < n) dst[i] = (__bf16)src[i];
}

// ---------------------------------------------------------------------------
// 3) Normalize (optional) + transpose to X^T[n][c] bf16, LDS 32x32 tiles.
//    z = b*3 + which : 0=content-norm, 1=style-norm, 2=style-raw
// ---------------------------------------------------------------------------
__global__ __launch_bounds__(256) void norm_transpose_kernel(
    const float* __restrict__ content, const float* __restrict__ style,
    const float* __restrict__ stats,
    __bf16* __restrict__ CnT, __bf16* __restrict__ SnT, __bf16* __restrict__ StT)
{
    const int z = blockIdx.z, b = z / 3, which = z - b * 3;
    const float* src = (which == 0) ? content : style;
    __bf16* dst = (which == 0) ? CnT : (which == 1 ? SnT : StT);
    const int srcsel = (which == 0) ? 0 : 1;
    const int n0 = blockIdx.x * 32, c0 = blockIdx.y * 32;
    const int t = threadIdx.x;
    __shared__ __bf16 tile[32][33];

    for (int i = 0; i < 4; i++) {
        int idx = t + i * 256, cr = idx >> 5, nc = idx & 31;
        int c = c0 + cr;
        float x = src[((size_t)b * Cc + c) * NSP + n0 + nc];
        if (which < 2) {
            float2 st = ((const float2*)stats)[(srcsel * Bb + b) * Cc + c];
            x = (x - st.x) * st.y;
        }
        tile[cr][nc] = (__bf16)x;
    }
    __syncthreads();
    for (int i = 0; i < 4; i++) {
        int idx = t + i * 256, nr = idx >> 5, cc = idx & 31;
        dst[((size_t)b * NSP + n0 + nr) * Cc + c0 + cc] = tile[cc][nr];
    }
}

// ---------------------------------------------------------------------------
// 4) Generic bf16 GEMM:  D stored [N][M] bf16.  K = 512.
//    Block tile 128x128, 8 waves, wave = 32(M)x64(N) = 8 accum tiles.
//    LDS rows stride 40 halfs (80 B, 16B-aligned). Async-to-LDS double
//    buffering when available, synchronous single-buffer fallback otherwise.
// ---------------------------------------------------------------------------
static __device__ __forceinline__ void gemm_compute_tile(
    const __bf16* As, const __bf16* Bs, int wm, int wn, int lo, int hi, v8f acc[2][4])
{
    Frag a[2], bf[4];
    for (int mt = 0; mt < 2; mt++) {
        int ar = (wm * 32 + mt * 16 + lo) * 40;
        a[mt].u[0] = *(const U4*)&As[ar + hi * 16];
        a[mt].u[1] = *(const U4*)&As[ar + hi * 16 + 8];
    }
    for (int nt = 0; nt < 4; nt++) {
        int br = (wn * 64 + nt * 16 + lo) * 40;
        bf[nt].u[0] = *(const U4*)&Bs[br + hi * 16];
        bf[nt].u[1] = *(const U4*)&Bs[br + hi * 16 + 8];
    }
    for (int mt = 0; mt < 2; mt++)
        for (int nt = 0; nt < 4; nt++)
            acc[mt][nt] = WMMA_BF16(a[mt].v, bf[nt].v, acc[mt][nt]);
}

__global__ __launch_bounds__(256) void gemm_bt_kernel(
    const __bf16* __restrict__ A,  long astride,
    const __bf16* __restrict__ Bt, long bstride,
    __bf16* __restrict__ Dt,       long dstride,
    const float* __restrict__ bias, int bias_mode, int relu,
    int M, int N)
{
    constexpr int K = 512;
    const __bf16* Ab = A  + (size_t)blockIdx.z * astride;
    const __bf16* Bv = Bt + (size_t)blockIdx.z * bstride;
    __bf16*       Db = Dt + (size_t)blockIdx.z * dstride;
    const int n0 = blockIdx.x * 128, m0 = blockIdx.y * 128;
    const int t = threadIdx.x, lane = t & 31, wave = t >> 5;
    const int wm = wave & 3, wn = wave >> 2;
    const int lo = lane & 15, hi = lane >> 4;

    const v8f vz = {};
    v8f acc[2][4];
    for (int i = 0; i < 2; i++) for (int j = 0; j < 4; j++) acc[i][j] = vz;

    const int lrow = t >> 1, lpart = t & 1;
    const __bf16* gA = Ab + (size_t)(m0 + lrow) * K + lpart * 16;
    const __bf16* gB = Bv + (size_t)(n0 + lrow) * K + lpart * 16;

#if USE_ASYNC_LDS
    __shared__ __bf16 Asm[2][128 * 40];
    __shared__ __bf16 Bsm[2][128 * 40];
    const int lloc = lrow * 40 + lpart * 16;

    // prologue: stage tile 0 into buffer 0
    ASYNC_B128(gA,     &Asm[0][lloc]);
    ASYNC_B128(gA + 8, &Asm[0][lloc + 8]);
    ASYNC_B128(gB,     &Bsm[0][lloc]);
    ASYNC_B128(gB + 8, &Bsm[0][lloc + 8]);

    for (int kt = 0; kt < K / 32; kt++) {
        const int cur = kt & 1;
        wait_async0();        // my tile kt has landed
        __syncthreads();      // everyone's tile kt landed; prior reads of buf cur^1 done
        if (kt + 1 < K / 32) {
            const __bf16* nA = gA + (size_t)(kt + 1) * 32;
            const __bf16* nB = gB + (size_t)(kt + 1) * 32;
            ASYNC_B128(nA,     &Asm[cur ^ 1][lloc]);
            ASYNC_B128(nA + 8, &Asm[cur ^ 1][lloc + 8]);
            ASYNC_B128(nB,     &Bsm[cur ^ 1][lloc]);
            ASYNC_B128(nB + 8, &Bsm[cur ^ 1][lloc + 8]);
        }
        gemm_compute_tile(&Asm[cur][0], &Bsm[cur][0], wm, wn, lo, hi, acc);
    }
#else
    __shared__ __bf16 Asm[128 * 40];
    __shared__ __bf16 Bsm[128 * 40];
    for (int kk = 0; kk < K; kk += 32) {
        U4 a0 = *(const U4*)(gA + kk);
        U4 a1 = *(const U4*)(gA + kk + 8);
        U4 b0 = *(const U4*)(gB + kk);
        U4 b1 = *(const U4*)(gB + kk + 8);
        __syncthreads();
        *(U4*)&Asm[lrow * 40 + lpart * 16]     = a0;
        *(U4*)&Asm[lrow * 40 + lpart * 16 + 8] = a1;
        *(U4*)&Bsm[lrow * 40 + lpart * 16]     = b0;
        *(U4*)&Bsm[lrow * 40 + lpart * 16 + 8] = b1;
        __syncthreads();
        gemm_compute_tile(Asm, Bsm, wm, wn, lo, hi, acc);
    }
#endif

    // Epilogue: bias (+relu), transposed contiguous b128 stores.
    for (int mt = 0; mt < 2; mt++) {
        for (int nt = 0; nt < 4; nt++) {
            int col  = n0 + wn * 64 + nt * 16 + lo;
            int rowb = m0 + wm * 32 + mt * 16 + hi * 8;
            float bn = (bias_mode == 1) ? bias[col] : 0.f;
            __align__(16) __bf16 tmp[8];
            for (int r = 0; r < 8; r++) {
                float v = acc[mt][nt][r] + ((bias_mode == 0) ? bias[rowb + r] : bn);
                if (relu) v = fmaxf(v, 0.f);
                tmp[r] = (__bf16)v;
            }
            *(U4*)(Db + (size_t)col * M + rowb) = *(const U4*)tmp;
        }
    }
}

// ---------------------------------------------------------------------------
// 5) Flash attention. Q^T[n][c], K^T[n][c], V[c][s] bf16; out styled^T[n][c].
//    32-query block, 64-key tiles, streaming softmax parallel across all
//    8 waves (4 rows/wave, 8 lanes/row, shuffle-xor reductions).
// ---------------------------------------------------------------------------
__global__ __launch_bounds__(256) void attention_kernel(
    const __bf16* __restrict__ Qt, const __bf16* __restrict__ Kt,
    const __bf16* __restrict__ Vn, __bf16* __restrict__ styledT)
{
    const int b = blockIdx.y, q0 = blockIdx.x * 32;
    const int t = threadIdx.x, lane = t & 31, wave = t >> 5;
    const int lo = lane & 15, hi = lane >> 4;
    const __bf16* Qb = Qt + (size_t)b * NSP * Cc;
    const __bf16* Kb = Kt + (size_t)b * NSP * Cc;
    const __bf16* Vb = Vn + (size_t)b * Cc * NSP;
    __bf16*       Ob = styledT + (size_t)b * NSP * Cc;

    __shared__ __bf16 Qs[32 * 520];    // Q tile [32q][512c]
    __shared__ float  Sbuf[32 * 68];   // score tile [32q][64s]
    __shared__ __bf16 Pbuf[32 * 80];   // exp(S-m) bf16, A-operand layout
    __shared__ float  mstate[32], lstate[32], alphaS[32];

    for (int j = 0; j < 8; j++) {      // stage Q block (32x512 bf16) once
        int u = t + j * 256, row = u >> 6, col8 = u & 63;
        *(U4*)&Qs[row * 520 + col8 * 8] = *(const U4*)&Qb[(size_t)(q0 + row) * Cc + col8 * 8];
    }
    if (t < 32) { mstate[t] = -3.0e38f; lstate[t] = 0.f; }
    const v8f vz = {};
    v8f acc[2][4];
    for (int i = 0; i < 2; i++) for (int j = 0; j < 4; j++) acc[i][j] = vz;
    __syncthreads();

    const int qm_s = wave & 1, sn_s = wave >> 1;

    for (int s0 = 0; s0 < NSP; s0 += 64) {
        // prefetch next key/value tiles toward L2/L0 while we compute
        if (s0 + 64 < NSP) {
            __builtin_prefetch(&Kb[(size_t)(s0 + 64 + sn_s * 16 + lo) * Cc + hi * 16], 0, 3);
            __builtin_prefetch(&Vb[(size_t)(wave * 64 + lo) * NSP + s0 + 64 + hi * 16], 0, 3);
        }

        // ---- scores: S[32,64] = Q_blk^T K_tile, K-dim = 512 ----
        v8f sc = vz;
        for (int kk = 0; kk < Cc; kk += 32) {
            Frag a, bf;
            int ar = (qm_s * 16 + lo) * 520 + kk;
            a.u[0] = *(const U4*)&Qs[ar + hi * 16];
            a.u[1] = *(const U4*)&Qs[ar + hi * 16 + 8];
            size_t kb = (size_t)(s0 + sn_s * 16 + lo) * Cc + kk + hi * 16;
            bf.u[0] = *(const U4*)&Kb[kb];
            bf.u[1] = *(const U4*)&Kb[kb + 8];
            sc = WMMA_BF16(a.v, bf.v, sc);
        }
        {
            int rb = qm_s * 16 + hi * 8, cb = sn_s * 16 + lo;
            for (int r = 0; r < 8; r++) Sbuf[(rb + r) * 68 + cb] = sc[r];
        }
        __syncthreads();

        // ---- streaming softmax: wave w owns rows 4w..4w+3, 8 lanes/row ----
        {
            const int row = wave * 4 + (lane >> 3);
            const int j0  = (lane & 7) * 8;
            float v[8];
            float mx = -3.0e38f;
            for (int k = 0; k < 8; k++) {
                v[k] = Sbuf[row * 68 + j0 + k];
                mx = fmaxf(mx, v[k]);
            }
            for (int m = 1; m < 8; m <<= 1) mx = fmaxf(mx, __shfl_xor(mx, m, 32));
            float mold = mstate[row];
            float mnew = fmaxf(mold, mx);
            float al = __expf(mold - mnew);
            float ls = 0.f;
            __align__(16) __bf16 pk[8];
            for (int k = 0; k < 8; k++) {
                float p = __expf(v[k] - mnew);
                ls += p;
                pk[k] = (__bf16)p;
            }
            *(U4*)&Pbuf[row * 80 + j0] = *(const U4*)pk;
            for (int m = 1; m < 8; m <<= 1) ls += __shfl_xor(ls, m, 32);
            if ((lane & 7) == 0) {
                mstate[row] = mnew;
                lstate[row] = lstate[row] * al + ls;
                alphaS[row] = al;
            }
        }
        __syncthreads();

        // ---- rescale accumulators ----
        for (int qm = 0; qm < 2; qm++) {
            float av[8];
            for (int r = 0; r < 8; r++) av[r] = alphaS[qm * 16 + hi * 8 + r];
            for (int cn = 0; cn < 4; cn++)
                for (int r = 0; r < 8; r++) acc[qm][cn][r] *= av[r];
        }

        // ---- acc += P[32,64] * V_tile^T  (K-dim = 64) ----
        for (int ks = 0; ks < 64; ks += 32) {
            Frag aP[2];
            for (int qm = 0; qm < 2; qm++) {
                int ar = (qm * 16 + lo) * 80 + ks;
                aP[qm].u[0] = *(const U4*)&Pbuf[ar + hi * 16];
                aP[qm].u[1] = *(const U4*)&Pbuf[ar + hi * 16 + 8];
            }
            for (int cn = 0; cn < 4; cn++) {
                Frag bv;
                size_t va = (size_t)(wave * 64 + cn * 16 + lo) * NSP + s0 + ks + hi * 16;
                bv.u[0] = *(const U4*)&Vb[va];
                bv.u[1] = *(const U4*)&Vb[va + 8];
                for (int qm = 0; qm < 2; qm++)
                    acc[qm][cn] = WMMA_BF16(aP[qm].v, bv.v, acc[qm][cn]);
            }
        }
        __syncthreads();
    }

    // ---- finalize: styled^T[q][c] = acc / l ----
    for (int qm = 0; qm < 2; qm++) {
        float inv[8];
        for (int r = 0; r < 8; r++) inv[r] = 1.f / lstate[qm * 16 + hi * 8 + r];
        for (int cn = 0; cn < 4; cn++) {
            int c = wave * 64 + cn * 16 + lo;
            for (int r = 0; r < 8; r++) {
                int q = q0 + qm * 16 + hi * 8 + r;
                Ob[(size_t)q * Cc + c] = (__bf16)(acc[qm][cn][r] * inv[r]);
            }
        }
    }
}

// ---------------------------------------------------------------------------
// 6) Final: out = (inst_norm(content)*sw + sb) * gamma + beta.
//    gamma/beta arrive transposed [n][c]; LDS 32x32 transpose.
// ---------------------------------------------------------------------------
__global__ __launch_bounds__(256) void final_kernel(
    const float* __restrict__ content, const float* __restrict__ stats,
    const __bf16* __restrict__ GT, const __bf16* __restrict__ BT,
    const float* __restrict__ sw, const float* __restrict__ sb,
    float* __restrict__ out)
{
    const int b = blockIdx.z, n0 = blockIdx.x * 32, c0 = blockIdx.y * 32;
    const int t = threadIdx.x;
    __shared__ float gs[32][33], bs[32][33];

    for (int i = 0; i < 4; i++) {
        int idx = t + i * 256, nr = idx >> 5, cc = idx & 31;
        size_t o = ((size_t)b * NSP + n0 + nr) * Cc + c0 + cc;
        gs[cc][nr] = (float)GT[o];
        bs[cc][nr] = (float)BT[o];
    }
    __syncthreads();
    for (int i = 0; i < 4; i++) {
        int idx = t + i * 256, cr = idx >> 5, nc = idx & 31;
        int c = c0 + cr;
        float2 st = ((const float2*)stats)[(0 * Bb + b) * Cc + c];  // content stats
        size_t o = ((size_t)b * Cc + c) * NSP + n0 + nc;
        float xn = (content[o] - st.x) * st.y;
        float sh = xn * sw[c] + sb[c];
        out[o] = sh * gs[cr][nc] + bs[cr][nc];
    }
}

// ---------------------------------------------------------------------------
// Host launcher
// ---------------------------------------------------------------------------
extern "C" void kernel_launch(void* const* d_in, const int* in_sizes, int n_in,
                              void* d_out, int out_size, void* d_ws, size_t ws_size,
                              hipStream_t stream)
{
    const float* content = (const float*)d_in[0];
    const float* style   = (const float*)d_in[1];
    const float* Wq = (const float*)d_in[2];  const float* bq = (const float*)d_in[3];
    const float* Wk = (const float*)d_in[4];  const float* bk = (const float*)d_in[5];
    const float* Wv = (const float*)d_in[6];  const float* bv = (const float*)d_in[7];
    const float* Wg = (const float*)d_in[8];  const float* bg = (const float*)d_in[9];
    const float* Wb = (const float*)d_in[10]; const float* bb = (const float*)d_in[11];
    const float* sw = (const float*)d_in[12]; const float* sb = (const float*)d_in[13];
    float* out = (float*)d_out;

    // workspace layout
    const size_t WELEM = (size_t)Cc * Cc;          // 262144
    const size_t XELEM = (size_t)Bb * NSP * Cc;    // 8388608
    char* W = (char*)d_ws;
    float*  stats = (float*)W;                     // 2*B*C float2 = 32 KB
    __bf16* WBq = (__bf16*)(W + 32768);
    __bf16* WBk = WBq + WELEM;
    __bf16* WBv = WBk + WELEM;
    __bf16* WBg = WBv + WELEM;
    __bf16* WBb = WBg + WELEM;
    __bf16* CnT = WBb + WELEM;     // content normalized, transposed [b][n][c]
    __bf16* SnT = CnT + XELEM;     // style normalized, transposed
    __bf16* StT = SnT + XELEM;     // style raw, transposed
    __bf16* Qt  = StT + XELEM;     // Q^T [b][n][c]
    __bf16* Kt  = Qt + XELEM;      // K^T [b][n][c]
    __bf16* Vn  = Kt + XELEM;      // V   [b][c][n]
    __bf16* styT = CnT;            // styled^T  (aliases CnT, free after Q GEMM)
    __bf16* GT   = SnT;            // gamma^T   (aliases SnT, free after K GEMM)
    __bf16* BT   = StT;            // beta^T    (aliases StT, free after V GEMM)

    const long XS = (long)NSP * Cc;   // per-batch stride (elements)

    stats_kernel<<<dim3(Cc, Bb, 2), 256, 0, stream>>>(content, style, stats);

    cvt_bf16_kernel<<<1024, 256, 0, stream>>>(Wq, WBq, (int)WELEM);
    cvt_bf16_kernel<<<1024, 256, 0, stream>>>(Wk, WBk, (int)WELEM);
    cvt_bf16_kernel<<<1024, 256, 0, stream>>>(Wv, WBv, (int)WELEM);
    cvt_bf16_kernel<<<1024, 256, 0, stream>>>(Wg, WBg, (int)WELEM);
    cvt_bf16_kernel<<<1024, 256, 0, stream>>>(Wb, WBb, (int)WELEM);

    norm_transpose_kernel<<<dim3(NSP / 32, Cc / 32, Bb * 3), 256, 0, stream>>>(
        content, style, stats, CnT, SnT, StT);

    // Q^T = (Wq * Cn)^T ; K^T = (Wk * Sn)^T : M=512(o), N=4096(n)
    gemm_bt_kernel<<<dim3(NSP / 128, Cc / 128, Bb), 256, 0, stream>>>(
        WBq, 0, CnT, XS, Qt, XS, bq, 0, 0, Cc, NSP);
    gemm_bt_kernel<<<dim3(NSP / 128, Cc / 128, Bb), 256, 0, stream>>>(
        WBk, 0, SnT, XS, Kt, XS, bk, 0, 0, Cc, NSP);
    // V[c][n] = (St^T * Wv^T)^T : M=4096(n), N=512(o), bias per N-col
    gemm_bt_kernel<<<dim3(Cc / 128, NSP / 128, Bb), 256, 0, stream>>>(
        StT, XS, WBv, 0, Vn, XS, bv, 1, 0, NSP, Cc);

    attention_kernel<<<dim3(NSP / 32, Bb), 256, 0, stream>>>(Qt, Kt, Vn, styT);

    // gamma^T / beta^T = relu(W * styled)^T
    gemm_bt_kernel<<<dim3(NSP / 128, Cc / 128, Bb), 256, 0, stream>>>(
        WBg, 0, styT, XS, GT, XS, bg, 0, 1, Cc, NSP);
    gemm_bt_kernel<<<dim3(NSP / 128, Cc / 128, Bb), 256, 0, stream>>>(
        WBb, 0, styT, XS, BT, XS, bb, 0, 1, Cc, NSP);

    final_kernel<<<dim3(NSP / 32, Cc / 32, Bb), 256, 0, stream>>>(
        content, stats, GT, BT, sw, sb, out);
}
